// InterScaleAttention_45689862095563
// MI455X (gfx1250) — matrix operand order repaired and checked by reading, not compiled, not run on hardware
//
#include <hip/hip_runtime.h>

// ---------------------------------------------------------------------------
// MI455X (gfx1250) fused window-attention, f16 WMMA (16x16x32, f32 accum).
// ~103 GFLOP vs ~260MB traffic -> compute bound -> run all three GEMMs on
// v_wmma_f32_16x16x32_f16. Double-buffered LDS in the GEMMs; async
// global->LDS copy (ASYNCcnt) for the attention K tile.
// ---------------------------------------------------------------------------

typedef __attribute__((ext_vector_type(16))) _Float16 v16h;
typedef __attribute__((ext_vector_type(8)))  _Float16 v8h;
typedef __attribute__((ext_vector_type(8)))  float    v8f;
typedef __attribute__((ext_vector_type(4)))  float    v4f;

union H16 { _Float16 e[16]; v8h h[2]; v16h v; };   // one WMMA A/B fragment
union H64 { _Float16 e[64]; v8h h[8]; };

__device__ __forceinline__ v8f wmma_f16(v16h a, v16h b, v8f c) {
  // (neg_a, A, neg_b, B, c_mod, C, reuse_a, reuse_b)
  return __builtin_amdgcn_wmma_f32_16x16x32_f16(false, a, false, b,
                                                (short)0, c, false, false);
}

// CDNA5 async copy: LDS[lds_off] <= MEM[gaddr], 16B per lane, ASYNCcnt.
__device__ __forceinline__ void async_load_b128(unsigned lds_off, const void* gaddr) {
  asm volatile("global_load_async_to_lds_b128 %0, %1, off"
               :: "v"(lds_off), "v"(gaddr) : "memory");
}
__device__ __forceinline__ void wait_asynccnt0() {
  asm volatile("s_wait_asynccnt 0x0" ::: "memory");
}

// ---------------------------------------------------------------------------
// Batched GEMM:  Out[w] = A[w] (MxK) x B (KxN, shared) + bias
// Block tile 128x128, K-step 32, 8 wave32 waves, double-buffered LDS.
// ---------------------------------------------------------------------------
template<bool A_F16, bool OUT_F16>
__global__ __launch_bounds__(256)
void gemm_wmma_kernel(const void* __restrict__ Aany,
                      const float* __restrict__ B,
                      const float* __restrict__ bias,
                      void* __restrict__ Out,
                      int M, int N, int K)
{
  // pitch 40 halfs = 80B = 5*16B -> all fragment ds_loads are b128-aligned
  __shared__ _Float16 As[2][128 * 40];
  __shared__ _Float16 Bt[2][128 * 40];   // B tile transposed [n][k]

  const int n0 = blockIdx.x * 128;
  const int m0 = blockIdx.y * 128;
  const int w  = blockIdx.z;

  const int t    = threadIdx.x;
  const int lane = t & 31;
  const int wv   = t >> 5;
  const int wm   = wv & 3;        // wave M block (32 rows)
  const int wn   = wv >> 2;       // wave N block (64 cols)
  const int lr   = lane & 15;
  const int lh   = lane >> 4;

  const float*    Af = (const float*)   Aany + (size_t)w * M * K;
  const _Float16* Ah = (const _Float16*)Aany + (size_t)w * M * K;

  v8f acc[2][4];
  v8f zero = {};
#pragma unroll
  for (int i = 0; i < 2; ++i)
#pragma unroll
    for (int j = 0; j < 4; ++j) acc[i][j] = zero;

  // cooperative-load work split
  const int a_m = t >> 1;            // 0..127 : A row
  const int a_h = t & 1;             // 0/1    : A 16-half segment
  const int b_k = t & 31;            // 0..31  : B k-row
  const int b_c = (t >> 5) * 16;     // col segment (16 cols)

  v4f rawAf[4]; v8h rawAh[2]; v4f rawB[4];

  auto loadGlobal = [&](int ks) {
    if (A_F16) {
      const v8h* p = (const v8h*)(Ah + (size_t)(m0 + a_m) * K + ks + a_h * 16);
      rawAh[0] = p[0]; rawAh[1] = p[1];
    } else {
      const v4f* p = (const v4f*)(Af + (size_t)(m0 + a_m) * K + ks + a_h * 16);
#pragma unroll
      for (int i = 0; i < 4; ++i) rawAf[i] = p[i];
    }
    const v4f* q = (const v4f*)(B + (size_t)(ks + b_k) * N + n0 + b_c);
#pragma unroll
    for (int i = 0; i < 4; ++i) rawB[i] = q[i];
  };

  auto storeLds = [&](int buf) {
    H16 tmp;
    if (A_F16) { tmp.h[0] = rawAh[0]; tmp.h[1] = rawAh[1]; }
    else {
#pragma unroll
      for (int i = 0; i < 4; ++i)
#pragma unroll
        for (int c = 0; c < 4; ++c) tmp.e[i * 4 + c] = (_Float16)rawAf[i][c];
    }
    v8h* d = (v8h*)&As[buf][a_m * 40 + a_h * 16];
    d[0] = tmp.h[0]; d[1] = tmp.h[1];
#pragma unroll
    for (int i = 0; i < 4; ++i)
#pragma unroll
      for (int c = 0; c < 4; ++c)
        Bt[buf][(b_c + i * 4 + c) * 40 + b_k] = (_Float16)rawB[i][c];
  };

  loadGlobal(0);
  storeLds(0);
  __syncthreads();

  int cur = 0;
  for (int ks = 0; ks < K; ks += 32) {
    const bool has_next = (ks + 32 < K);
    if (has_next) loadGlobal(ks + 32);     // overlap with WMMAs below

    H16 a[2];
#pragma unroll
    for (int i = 0; i < 2; ++i) {
      const int row = wm * 32 + i * 16 + lr;
      a[i].h[0] = *(const v8h*)&As[cur][row * 40 + lh * 8];
      a[i].h[1] = *(const v8h*)&As[cur][row * 40 + 16 + lh * 8];
    }
#pragma unroll
    for (int j = 0; j < 4; ++j) {
      H16 b;
      const int col = wn * 64 + j * 16 + lr;
      b.h[0] = *(const v8h*)&Bt[cur][col * 40 + lh * 8];
      b.h[1] = *(const v8h*)&Bt[cur][col * 40 + 16 + lh * 8];
#pragma unroll
      for (int i = 0; i < 2; ++i)
        acc[i][j] = wmma_f16(a[i].v, b.v, acc[i][j]);
    }

    if (has_next) storeLds(cur ^ 1);
    __syncthreads();
    cur ^= 1;
  }

  // epilogue: bias add + store
#pragma unroll
  for (int j = 0; j < 4; ++j) {
    const int n  = n0 + wn * 64 + j * 16 + lr;
    const float bs = bias[n];
#pragma unroll
    for (int i = 0; i < 2; ++i) {
      const int mb = m0 + wm * 32 + i * 16 + lh * 8;
#pragma unroll
      for (int r = 0; r < 8; ++r) {
        const float vo = acc[i][j][r] + bs;
        const size_t idx = (size_t)w * M * N + (size_t)(mb + r) * N + n;
        if (OUT_F16) ((_Float16*)Out)[idx] = (_Float16)vo;
        else         ((float*)Out)[idx]    = vo;
      }
    }
  }
}

// ---------------------------------------------------------------------------
// Attention: one workgroup per (window, head, 64-row Q block).
// Full S block [64x512] f32 in LDS, exact softmax (matches reference, no
// 1/sqrt(d) scale).  K tile arrives via async global->LDS (ASYNCcnt).
// Q fragments load straight from global (16B-aligned strips, no staging).
// LDS: Ss f32[64][516] | P f16[64][520] | reduce f32[512] | Ks[512][72]/Vt[64][520]
// ---------------------------------------------------------------------------
#define SS_OFF  0
#define P_OFF   132096
#define RED_OFF 198656
#define KV_OFF  200704
#define ATTN_LDS_BYTES 274432

__global__ __launch_bounds__(256)
void attn_wmma_kernel(const _Float16* __restrict__ qkv,   // [64][512][1536]
                      _Float16* __restrict__ attn_out)    // [64][512][512]
{
  extern __shared__ char smem[];
  float*    Ss   = (float*)(smem + SS_OFF);       // [64][516]
  _Float16* P    = (_Float16*)(smem + P_OFF);     // [64][520]
  float*    redM = (float*)(smem + RED_OFF);      // [256]
  float*    redS = (float*)(smem + RED_OFF + 1024);
  _Float16* Ks   = (_Float16*)(smem + KV_OFF);    // [512][72]
  _Float16* Vt   = (_Float16*)(smem + KV_OFF);    // [64][520] (after pass 1)

  const int blk = blockIdx.x;
  const int qb = blk & 7, h = (blk >> 3) & 7, w = blk >> 6;
  const int q0 = qb * 64;

  const int t    = threadIdx.x;
  const int lane = t & 31;
  const int wv   = t >> 5;
  const int lr   = lane & 15;
  const int lh   = lane >> 4;

  const size_t wbase = (size_t)w * 512 * 1536;

  // ---- async copy K [512][64] f16 -> Ks (pitch 72 halfs = 144B) -------
  {
    const char* kg = (const char*)(qkv + wbase + 512 + h * 64);
#pragma unroll
    for (int c = 0; c < 16; ++c) {
      const int task = c * 256 + t;            // 0..4095 lane-tasks
      const int key  = task >> 3;              // 0..511
      const int seg  = task & 7;               // 16B chunk within row
      async_load_b128((unsigned)(KV_OFF + key * 144 + seg * 16),
                      kg + (size_t)key * 3072 + seg * 16);
    }
  }
  wait_asynccnt0();
  __syncthreads();

  // ---- pass 1: S[64][512] = Q x K^T -----------------------------------
  const int mt = wv & 3;      // 16-row strip
  const int nh = wv >> 2;     // 256-col half
  {
    v8f s[16];
    v8f zero = {};
#pragma unroll
    for (int j = 0; j < 16; ++j) s[j] = zero;
#pragma unroll
    for (int dk = 0; dk < 64; dk += 32) {
      H16 a;   // A fragment straight from global (contiguous 16B strips)
      const int grow = q0 + mt * 16 + lr;
      const _Float16* ga = qkv + wbase + (size_t)grow * 1536 + h * 64 + dk;
      a.h[0] = *(const v8h*)(ga + lh * 8);
      a.h[1] = *(const v8h*)(ga + 16 + lh * 8);
#pragma unroll
      for (int j = 0; j < 16; ++j) {
        H16 b;
        const int key = nh * 256 + j * 16 + lr;
        b.h[0] = *(const v8h*)&Ks[key * 72 + dk + lh * 8];
        b.h[1] = *(const v8h*)&Ks[key * 72 + dk + 16 + lh * 8];
        s[j] = wmma_f16(a.v, b.v, s[j]);
      }
    }
#pragma unroll
    for (int j = 0; j < 16; ++j) {
      const int col = nh * 256 + j * 16 + lr;
      const int rb  = mt * 16 + lh * 8;
#pragma unroll
      for (int r = 0; r < 8; ++r) Ss[(rb + r) * 516 + col] = s[j][r];
    }
  }
  __syncthreads();

  // ---- load V transposed [64 d][512 key] (overwrites K buffer) --------
#pragma unroll
  for (int kk = 0; kk < 2; ++kk) {
    const int key = t * 2 + kk;
    const v8h* src = (const v8h*)(qkv + wbase + (size_t)key * 1536 + 1024 + h * 64);
    H64 tmp;
#pragma unroll
    for (int i = 0; i < 8; ++i) tmp.h[i] = src[i];
#pragma unroll
    for (int d = 0; d < 64; ++d) Vt[d * 520 + key] = tmp.e[d];
  }

  // ---- exact softmax, 4 threads per row -------------------------------
  const int srow = t >> 2;            // 0..63
  const int scol = (t & 3) * 128;     // 128-col segment
  {
    float* row = &Ss[srow * 516];
    float lmax = -3.402823466e38f;
    for (int c = scol; c < scol + 128; ++c) lmax = fmaxf(lmax, row[c]);
    redM[t] = lmax;
  }
  __syncthreads();
  {
    float* row = &Ss[srow * 516];
    const float rmax = fmaxf(fmaxf(redM[srow * 4 + 0], redM[srow * 4 + 1]),
                             fmaxf(redM[srow * 4 + 2], redM[srow * 4 + 3]));
    float lsum = 0.f;
    for (int c = scol; c < scol + 128; ++c) {
      const float e = __expf(row[c] - rmax);
      row[c] = e;
      lsum += e;
    }
    redS[t] = lsum;
  }
  __syncthreads();
  {
    float* row = &Ss[srow * 516];
    const float inv = 1.0f / (redS[srow * 4 + 0] + redS[srow * 4 + 1] +
                              redS[srow * 4 + 2] + redS[srow * 4 + 3]);
    _Float16* prow = &P[srow * 520];
    for (int c = scol; c < scol + 128; ++c) prow[c] = (_Float16)(row[c] * inv);
  }
  __syncthreads();

  // ---- pass 2: O[64][64] = P[64][512] x V[512][64] --------------------
  {
    const int mt2 = wv & 3;
    const int np  = wv >> 2;           // n-tiles np*2, np*2+1
    v8f o[2];
    v8f zero = {};
    o[0] = zero; o[1] = zero;
    for (int ks = 0; ks < 512; ks += 32) {
      H16 a;
      const int row = mt2 * 16 + lr;
      a.h[0] = *(const v8h*)&P[row * 520 + ks + lh * 8];
      a.h[1] = *(const v8h*)&P[row * 520 + ks + 16 + lh * 8];
#pragma unroll
      for (int j = 0; j < 2; ++j) {
        H16 b;
        const int d = (np * 2 + j) * 16 + lr;
        b.h[0] = *(const v8h*)&Vt[d * 520 + ks + lh * 8];
        b.h[1] = *(const v8h*)&Vt[d * 520 + ks + 16 + lh * 8];
        o[j] = wmma_f16(a.v, b.v, o[j]);
      }
    }
#pragma unroll
    for (int j = 0; j < 2; ++j) {
      const int d  = (np * 2 + j) * 16 + lr;
      const int rb = mt2 * 16 + lh * 8;
#pragma unroll
      for (int r = 0; r < 8; ++r)
        attn_out[(size_t)w * 512 * 512 + (size_t)(q0 + rb + r) * 512 + h * 64 + d]
            = (_Float16)o[j][r];
    }
  }
}

// ---------------------------------------------------------------------------
extern "C" void kernel_launch(void* const* d_in, const int* in_sizes, int n_in,
                              void* d_out, int out_size, void* d_ws, size_t ws_size,
                              hipStream_t stream) {
  (void)in_sizes; (void)n_in; (void)out_size; (void)ws_size;

  const float* x      = (const float*)d_in[0];   // [64][512][512]
  const float* w_qkv  = (const float*)d_in[1];   // [512][1536]
  const float* b_qkv  = (const float*)d_in[2];   // [1536]
  const float* w_proj = (const float*)d_in[3];   // [512][512]
  const float* b_proj = (const float*)d_in[4];   // [512]
  float* out = (float*)d_out;                    // [64][512][512]

  // workspace: qkv f16 (96MB) + attention output f16 (32MB) = 128MB
  _Float16* qkv_h  = (_Float16*)d_ws;                                  // 64*512*1536
  _Float16* attn_h = (_Float16*)((char*)d_ws + (size_t)100663296);     // 64*512*512

  // Stage 1: qkv = x @ w_qkv + b_qkv   (f32 in, f16 out)
  gemm_wmma_kernel<false, true>
      <<<dim3(1536 / 128, 512 / 128, 64), 256, 0, stream>>>(
          x, w_qkv, b_qkv, qkv_h, 512, 1536, 512);

  // Stage 2: per (window, head, qblock) exact-softmax attention
  attn_wmma_kernel<<<dim3(64 * 8 * 8), 256, ATTN_LDS_BYTES, stream>>>(qkv_h, attn_h);

  // Stage 3: out = attn @ w_proj + b_proj   (f16 in, f32 out)
  gemm_wmma_kernel<true, false>
      <<<dim3(512 / 128, 512 / 128, 64), 256, 0, stream>>>(
          attn_h, w_proj, b_proj, out, 512, 512, 512);
}
